// FusedMoE_6365141532678
// MI455X (gfx1250) — matrix operand order repaired and checked by reading, not compiled, not run on hardware
//
#include <hip/hip_runtime.h>
#include <hip/hip_bf16.h>

// ---------------- problem constants ----------------
static constexpr int T = 8192;
static constexpr int H = 2048;
static constexpr int F = 1408;
static constexpr int E = 8;

static constexpr int LDSP = 40;   // LDS row pitch in bf16 elements (32 data + 4-DWORD TDM pad)

// ---------------- WMMA types ----------------
typedef __attribute__((ext_vector_type(16))) __bf16 v16bf;
typedef __attribute__((ext_vector_type(8)))  float  v8f;
typedef __attribute__((ext_vector_type(8)))  int    v8i;

// TDM descriptor groups (clang-23 6-arg builtin form:
//   (uint32x4 g0, int32x8 g1, int32x4 g2, int32x4 g3, int32x8 g4, int cpol))
typedef __attribute__((ext_vector_type(4))) unsigned int tdm_g0_t;
typedef __attribute__((ext_vector_type(8))) int          tdm_g1_t;
typedef __attribute__((ext_vector_type(4))) int          tdm_g23_t;
typedef __attribute__((ext_vector_type(8))) int          tdm_g4_t;

union FragCast { v8i i; v16bf b; };
union FragLd   { struct { uint4 lo, hi; } u; v8i v; };

__device__ inline v8f wmma_bf16(v8i a, v8i b, v8f c) {
  FragCast A; A.i = a;
  FragCast B; B.i = b;
  return __builtin_amdgcn_wmma_f32_16x16x32_bf16(
      false, A.b, false, B.b, (short)0, c, false, false);
}

// A-matrix 16x32 bf16 fragment (ISA 7.12.2):
// lane<16 : VGPR0-3 = K0..7,  VGPR4-7 = K16..23   (row M=lane&15)
// lane>=16: VGPR0-3 = K8..15, VGPR4-7 = K24..31
__device__ inline v8i load_frag_A(const unsigned short* __restrict__ row, int k0, int lane) {
  int koff = (lane & 16) ? 8 : 0;
  FragLd f;
  f.u.lo = *reinterpret_cast<const uint4*>(row + k0 + koff);
  f.u.hi = *reinterpret_cast<const uint4*>(row + k0 + koff + 16);
  return f.v;
}

// B-matrix 32x16 bf16 fragment from an LDS tile with row pitch LDSP:
// lane<16 : VGPR0-3 = K0..7, VGPR4-7 = K16..23 (col N=lane&15); lane>=16: K8..15, K24..31
__device__ inline v8i load_frag_B_lds(const unsigned short* tile, int row, int lane) {
  int koff = (lane & 16) ? 8 : 0;
  const unsigned short* p = tile + row * LDSP + koff;
  FragLd f;
  f.u.lo = *reinterpret_cast<const uint4*>(p);
  f.u.hi = *reinterpret_cast<const uint4*>(p + 16);
  return f.v;
}

__device__ inline unsigned short f2bf(float f) {
  unsigned u = __float_as_uint(f);
  u += 0x7fffu + ((u >> 16) & 1u);            // round-to-nearest-even
  return (unsigned short)(u >> 16);
}

__device__ inline void atomAddF32(float* p, float v) {
  __hip_atomic_fetch_add(p, v, __ATOMIC_RELAXED, __HIP_MEMORY_SCOPE_AGENT);
}

// ---------------- Tensor Data Mover: 2D tile (rows x 32 bf16) -> LDS ----------------
// D# per CDNA5 ISA ch. 8: group0 = {flags, lds_addr, global_addr, type=2},
// group1 = {data_size=2B + LDS pad cfg, tensor dims, tile dims, dim0 stride}.
// Padding: 4 DWORDs after every 16 DWORDs stored -> LDS row pitch 80B (= LDSP elems).
__device__ inline void tdm_load_2d(unsigned lds_off, const unsigned short* gptr,
                                   unsigned tensor_d0, unsigned tensor_d1,
                                   unsigned tile_d1, unsigned stride_elems) {
  unsigned long long ga = (unsigned long long)(uintptr_t)gptr;
  tdm_g0_t g0;
  g0[0] = 1u;                                        // count=1, user descriptor
  g0[1] = lds_off;                                   // lds_addr
  g0[2] = (unsigned)(ga & 0xffffffffu);              // global_addr[31:0]
  g0[3] = (unsigned)((ga >> 32) & 0x01ffffffu) | (2u << 30);  // addr[56:32] | type=2
  tdm_g1_t g1;
  g1[0] = (int)((1u << 16)      // data_size = 2 bytes
              | (1u << 20)      // pad_enable
              | (3u << 22)      // pad_interval: 16 DWORDs
              | (3u << 25));    // pad_amount: 4 DWORDs
  g1[1] = (int)((tensor_d0 & 0xffffu) << 16);                          // dim0[15:0]
  g1[2] = (int)(((tensor_d0 >> 16) & 0xffffu) | ((tensor_d1 & 0xffffu) << 16));
  g1[3] = (int)(((tensor_d1 >> 16) & 0xffffu) | (32u << 16));          // tile_dim0 = 32
  g1[4] = (int)(tile_d1 & 0xffffu);                                    // tile_dim1 (tile_dim2 = 0)
  g1[5] = (int)stride_elems;                                           // dim0 stride low 32
  g1[6] = 0;
  g1[7] = 0;
  tdm_g23_t z4 = {0, 0, 0, 0};
  tdm_g4_t  z8 = {0, 0, 0, 0, 0, 0, 0, 0};
  __builtin_amdgcn_tensor_load_to_lds(g0, g1, z4, z4, z8, 0);
}

// ---------------- f32 -> bf16 bulk convert (4-wide) ----------------
__global__ void cvt_f32_bf16(const float* __restrict__ s, unsigned short* __restrict__ d,
                             long long n4) {
  long long i = (long long)blockIdx.x * blockDim.x + threadIdx.x;
  long long stride = (long long)gridDim.x * blockDim.x;
  for (; i < n4; i += stride) {
    float4 v = reinterpret_cast<const float4*>(s)[i];
    ushort4 o;
    o.x = f2bf(v.x); o.y = f2bf(v.y); o.z = f2bf(v.z); o.w = f2bf(v.w);
    reinterpret_cast<ushort4*>(d)[i] = o;
  }
}

// ---------------- routing ----------------
__global__ void route_count(const float* __restrict__ logits,
                            int* __restrict__ tk_e, float* __restrict__ tk_w,
                            int* __restrict__ cnt) {
  int t = blockIdx.x * blockDim.x + threadIdx.x;
  if (t >= T) return;
  float l[E];
#pragma unroll
  for (int e = 0; e < E; ++e) l[e] = logits[t * E + e];
  int i0 = 0;
#pragma unroll
  for (int e = 1; e < E; ++e) if (l[e] > l[i0]) i0 = e;
  int i1 = (i0 == 0) ? 1 : 0;
#pragma unroll
  for (int e = 0; e < E; ++e) if (e != i0 && l[e] > l[i1]) i1 = e;
  float d = expf(l[i1] - l[i0]);
  float w0 = 1.0f / (1.0f + d);
  float w1 = 1.0f - w0;
  tk_e[2 * t] = i0; tk_e[2 * t + 1] = i1;
  tk_w[2 * t] = w0; tk_w[2 * t + 1] = w1;
  atomicAdd(&cnt[i0], 1);
  atomicAdd(&cnt[i1], 1);
}

__global__ void route_offsets(const int* __restrict__ cnt, int* __restrict__ offs) {
  if (threadIdx.x == 0 && blockIdx.x == 0) {
    int a = 0;
    for (int e = 0; e < E; ++e) { offs[e] = a; a += cnt[e]; }
  }
}

__global__ void route_scatter(const int* __restrict__ tk_e, const float* __restrict__ tk_w,
                              const int* __restrict__ offs,
                              int* __restrict__ tokIdx, float* __restrict__ wgt) {
  __shared__ int scan[256];
  __shared__ int base_s;
  int e = blockIdx.x;
  int tid = threadIdx.x;
  if (tid == 0) base_s = offs[e];
  __syncthreads();
  for (int c0 = 0; c0 < T; c0 += 256) {
    int t = c0 + tid;
    int p = 0; float w = 0.0f;
    if (tk_e[2 * t] == e)          { p = 1; w = tk_w[2 * t]; }
    else if (tk_e[2 * t + 1] == e) { p = 1; w = tk_w[2 * t + 1]; }
    scan[tid] = p;
    __syncthreads();
    for (int o = 1; o < 256; o <<= 1) {
      int v = (tid >= o) ? scan[tid - o] : 0;
      __syncthreads();
      scan[tid] += v;
      __syncthreads();
    }
    if (p) {
      int pos = base_s + scan[tid] - 1;
      tokIdx[pos] = t;
      wgt[pos]    = w;
    }
    __syncthreads();
    if (tid == 0) base_s += scan[255];
    __syncthreads();
  }
}

// ---------------- GEMM1: gate_up = X_e @ W1[e]^T, h = silu(gate)*up ----------------
// block 128(M) x 64(N); 8 waves = 4M x 2N; wave = 32x32 gate + 32x32 up.
// Weight tiles (gate 64x32, up 64x32) staged into LDS by TDM, double-buffered.
__global__ __launch_bounds__(256) void moe_gemm1(
    const unsigned short* __restrict__ xb,      // [T,H] bf16
    const unsigned short* __restrict__ w1b,     // [E,2F,H] bf16
    const int* __restrict__ tokIdx, const int* __restrict__ cnt,
    const int* __restrict__ offs,
    unsigned short* __restrict__ hb) {          // [2T,F] bf16
  __shared__ __align__(16) unsigned short sB[2][2][64 * LDSP];  // [buf][gate|up]
  int e  = blockIdx.z;
  int cn = cnt[e];
  int m0 = blockIdx.x * 128;
  if (m0 >= cn) return;
  int off_e = offs[e];

  int lane = threadIdx.x & 31;
  int wave = threadIdx.x >> 5;
  int wm = wave & 3, wn = wave >> 2;
  int mwb = m0 + wm * 32;
  int nblk = blockIdx.y * 64;
  int nwb = nblk + wn * 32;

  const unsigned short* we = w1b + (size_t)e * (2 * F) * H;
  const unsigned short* gateBase = we + (size_t)nblk * H;
  const unsigned short* upBase   = we + (size_t)(F + nblk) * H;

  const unsigned short* arow[2];
#pragma unroll
  for (int mt = 0; mt < 2; ++mt) {
    int rl = mwb + mt * 16 + (lane & 15);
    int tok = (rl < cn) ? tokIdx[off_e + rl] : tokIdx[off_e];
    arow[mt] = xb + (size_t)tok * H;
  }

  unsigned sbase = (unsigned)(uintptr_t)&sB[0][0][0];
  constexpr unsigned TILEB = 64 * LDSP * 2;   // bytes per tile

  constexpr int NK = H / 32;
  if (wave == 0) {
    tdm_load_2d(sbase + 0 * TILEB, gateBase + 0,  H, 2 * F, 64, H);
    tdm_load_2d(sbase + 1 * TILEB, upBase   + 0,  H, 2 * F, 64, H);
    tdm_load_2d(sbase + 2 * TILEB, gateBase + 32, H, 2 * F, 64, H);
    tdm_load_2d(sbase + 3 * TILEB, upBase   + 32, H, 2 * F, 64, H);
  }

  v8f cg[2][2] = {};
  v8f cu[2][2] = {};

  int rg0 = wn * 32 + (lane & 15);   // gate/up row inside 64-row LDS tile
  for (int i = 0; i < NK; ++i) {
    if (wave == 0) {
      if (i + 1 < NK) __builtin_amdgcn_s_wait_tensorcnt(2);  // oldest chunk landed
      else            __builtin_amdgcn_s_wait_tensorcnt(0);
    }
    __syncthreads();                                          // tile buf[i&1] ready
    int buf = i & 1;
    const unsigned short* sg = &sB[buf][0][0];
    const unsigned short* su = &sB[buf][1][0];
    int k0 = i * 32;

    v8i a0 = load_frag_A(arow[0], k0, lane);
    v8i a1 = load_frag_A(arow[1], k0, lane);
    v8i g0 = load_frag_B_lds(sg, rg0,      lane);
    v8i g1 = load_frag_B_lds(sg, rg0 + 16, lane);
    v8i u0 = load_frag_B_lds(su, rg0,      lane);
    v8i u1 = load_frag_B_lds(su, rg0 + 16, lane);
    cg[0][0] = wmma_bf16(a0, g0, cg[0][0]);
    cg[0][1] = wmma_bf16(a0, g1, cg[0][1]);
    cg[1][0] = wmma_bf16(a1, g0, cg[1][0]);
    cg[1][1] = wmma_bf16(a1, g1, cg[1][1]);
    cu[0][0] = wmma_bf16(a0, u0, cu[0][0]);
    cu[0][1] = wmma_bf16(a0, u1, cu[0][1]);
    cu[1][0] = wmma_bf16(a1, u0, cu[1][0]);
    cu[1][1] = wmma_bf16(a1, u1, cu[1][1]);

    __syncthreads();                                          // everyone done with buf
    if (wave == 0 && i + 2 < NK) {
      int kn = (i + 2) * 32;
      tdm_load_2d(sbase + (buf * 2 + 0) * TILEB, gateBase + kn, H, 2 * F, 64, H);
      tdm_load_2d(sbase + (buf * 2 + 1) * TILEB, upBase   + kn, H, 2 * F, 64, H);
    }
  }

  int lh = (lane & 16) ? 8 : 0;
  int lc = lane & 15;
#pragma unroll
  for (int mt = 0; mt < 2; ++mt)
#pragma unroll
    for (int nt = 0; nt < 2; ++nt)
#pragma unroll
      for (int r = 0; r < 8; ++r) {
        int rl = mwb + mt * 16 + r + lh;
        if (rl < cn) {
          float g = cg[mt][nt][r];
          float u = cu[mt][nt][r];
          float hval = (g / (1.0f + expf(-g))) * u;   // silu(g)*u
          int col = nwb + nt * 16 + lc;
          hb[(size_t)(off_e + rl) * F + col] = f2bf(hval);
        }
      }
}

// ---------------- GEMM2: y = H_e @ W2[e]^T, out[tok] += w * y ----------------
// block 128(M) x 128(N); 8 waves = 4M x 2N; wave = 32x64 (2x4 WMMA tiles).
// W2 tile (128 rows x 32 K) staged via TDM, double-buffered.
__global__ __launch_bounds__(256) void moe_gemm2(
    const unsigned short* __restrict__ hb,      // [2T,F] bf16
    const unsigned short* __restrict__ w2b,     // [E,H,F] bf16
    const int* __restrict__ tokIdx, const float* __restrict__ wgt,
    const int* __restrict__ cnt, const int* __restrict__ offs,
    float* __restrict__ out) {                  // [T,H] f32
  __shared__ __align__(16) unsigned short sB[2][128 * LDSP];
  int e  = blockIdx.z;
  int cn = cnt[e];
  int m0 = blockIdx.x * 128;
  if (m0 >= cn) return;
  int off_e = offs[e];

  int lane = threadIdx.x & 31;
  int wave = threadIdx.x >> 5;
  int wm = wave & 3, wn = wave >> 2;
  int mwb = m0 + wm * 32;
  int nblk = blockIdx.y * 128;
  int nwb = nblk + wn * 64;

  const unsigned short* we = w2b + (size_t)e * H * F;
  const unsigned short* bBase = we + (size_t)nblk * F;

  const unsigned short* arow[2];
#pragma unroll
  for (int mt = 0; mt < 2; ++mt) {
    int rl = mwb + mt * 16 + (lane & 15);
    int rli = (rl < cn) ? rl : (cn - 1);
    arow[mt] = hb + (size_t)(off_e + rli) * F;
  }

  unsigned sbase = (unsigned)(uintptr_t)&sB[0][0];
  constexpr unsigned TILEB = 128 * LDSP * 2;

  constexpr int NK = F / 32;
  if (wave == 0) {
    tdm_load_2d(sbase + 0 * TILEB, bBase + 0,  F, H, 128, F);
    tdm_load_2d(sbase + 1 * TILEB, bBase + 32, F, H, 128, F);
  }

  v8f acc[2][4] = {};
  int rb0 = wn * 64 + (lane & 15);
  for (int i = 0; i < NK; ++i) {
    if (wave == 0) {
      if (i + 1 < NK) __builtin_amdgcn_s_wait_tensorcnt(1);
      else            __builtin_amdgcn_s_wait_tensorcnt(0);
    }
    __syncthreads();
    int buf = i & 1;
    const unsigned short* sb = &sB[buf][0];
    int k0 = i * 32;

    v8i a0 = load_frag_A(arow[0], k0, lane);
    v8i a1 = load_frag_A(arow[1], k0, lane);
#pragma unroll
    for (int nt = 0; nt < 4; ++nt) {
      v8i b = load_frag_B_lds(sb, rb0 + nt * 16, lane);
      acc[0][nt] = wmma_bf16(a0, b, acc[0][nt]);
      acc[1][nt] = wmma_bf16(a1, b, acc[1][nt]);
    }

    __syncthreads();
    if (wave == 0 && i + 2 < NK) {
      tdm_load_2d(sbase + buf * TILEB, bBase + (i + 2) * 32, F, H, 128, F);
    }
  }

  int lh = (lane & 16) ? 8 : 0;
  int lc = lane & 15;
#pragma unroll
  for (int mt = 0; mt < 2; ++mt)
#pragma unroll
    for (int r = 0; r < 8; ++r) {
      int rl = mwb + mt * 16 + r + lh;
      if (rl < cn) {
        int idx = off_e + rl;
        int tok = tokIdx[idx];
        float wv = wgt[idx];
#pragma unroll
        for (int nt = 0; nt < 4; ++nt) {
          int col = nwb + nt * 16 + lc;
          atomAddF32(&out[(size_t)tok * H + col], wv * acc[mt][nt][r]);
        }
      }
    }
}

// ---------------- launch ----------------
extern "C" void kernel_launch(void* const* d_in, const int* in_sizes, int n_in,
                              void* d_out, int out_size, void* d_ws, size_t ws_size,
                              hipStream_t stream) {
  const float* x      = (const float*)d_in[0];   // (T,H)
  const float* logits = (const float*)d_in[1];   // (T,E)
  const float* w1     = (const float*)d_in[2];   // (E,2F,H)
  const float* w2     = (const float*)d_in[3];   // (E,H,F)
  float* out = (float*)d_out;

  char* ws = (char*)d_ws;
  size_t cur = 0;
  auto carve = [&](size_t bytes) { void* p = ws + cur; cur = (cur + bytes + 255) & ~(size_t)255; return p; };
  int*   cnt    = (int*)  carve(E * sizeof(int));
  int*   offs   = (int*)  carve(E * sizeof(int));
  int*   tk_e   = (int*)  carve((size_t)2 * T * sizeof(int));
  float* tk_w   = (float*)carve((size_t)2 * T * sizeof(float));
  int*   tokIdx = (int*)  carve((size_t)2 * T * sizeof(int));
  float* wgt    = (float*)carve((size_t)2 * T * sizeof(float));
  unsigned short* xb  = (unsigned short*)carve((size_t)T * H * 2);
  unsigned short* w1b = (unsigned short*)carve((size_t)E * 2 * F * H * 2);
  unsigned short* w2b = (unsigned short*)carve((size_t)E * H * F * 2);
  unsigned short* hb  = (unsigned short*)carve((size_t)2 * T * F * 2);

  (void)hipMemsetAsync(cnt, 0, E * sizeof(int), stream);
  (void)hipMemsetAsync(out, 0, (size_t)T * H * sizeof(float), stream);

  route_count<<<T / 256, 256, 0, stream>>>(logits, tk_e, tk_w, cnt);
  route_offsets<<<1, 32, 0, stream>>>(cnt, offs);
  route_scatter<<<E, 256, 0, stream>>>(tk_e, tk_w, offs, tokIdx, wgt);

  cvt_f32_bf16<<<2048, 256, 0, stream>>>(x,  xb,  (long long)T * H / 4);
  cvt_f32_bf16<<<4096, 256, 0, stream>>>(w1, w1b, (long long)E * 2 * F * H / 4);
  cvt_f32_bf16<<<4096, 256, 0, stream>>>(w2, w2b, (long long)E * H * F / 4);

  moe_gemm1<<<dim3(T / 128, F / 64,  E), 256, 0, stream>>>(xb, w1b, tokIdx, cnt, offs, hb);
  moe_gemm2<<<dim3(T / 128, H / 128, E), 256, 0, stream>>>(hb, w2b, tokIdx, wgt, cnt, offs, out);
}
